// CEABlock_FMAG0_65721589563865
// MI455X (gfx1250) — matrix operand order repaired and checked by reading, compiled, not run on hardware
//
#include <hip/hip_runtime.h>
#include <math.h>

typedef _Float16 half_t;
typedef __attribute__((ext_vector_type(16))) _Float16 v16h;
typedef __attribute__((ext_vector_type(8)))  _Float16 v8h;
typedef __attribute__((ext_vector_type(8)))  float    v8f;

#define D_MODEL 768
#define N_HEADS 12
#define HEAD_D  64
#define N_TOK   1025
#define BATCH   4
#define N_SEL   257   // 1 + 256

__device__ __forceinline__ float gelu_f(float x) {
    const float c = 0.7978845608028654f;
    float t = c * (x + 0.044715f * x * x * x);
    return 0.5f * x * (1.0f + tanhf(t));
}

// ---------------------------------------------------------------------------
// CDNA5 async global->LDS copy (ASYNCcnt-tracked), 16B per lane.
// LDS operand is the low 32 bits of the generic pointer (LDS aperture maps
// addr[31:0] to the wave-relative LDS offset).
// ---------------------------------------------------------------------------
__device__ __forceinline__ void async_copy_b128(const void* g, void* lds) {
    unsigned lo = (unsigned)(uintptr_t)lds;
    unsigned long long ga = (unsigned long long)(uintptr_t)g;
    asm volatile("global_load_async_to_lds_b128 %0, %1, off" :: "v"(lo), "v"(ga) : "memory");
}
__device__ __forceinline__ void async_wait0() {
    asm volatile("s_wait_asynccnt 0x0" ::: "memory");
}

// ---------------------------------------------------------------------------
// fp32 -> fp16 conversion (weights)
// ---------------------------------------------------------------------------
__global__ void f32_to_f16_kernel(const float* __restrict__ a, half_t* __restrict__ o, int n) {
    int i = blockIdx.x * blockDim.x + threadIdx.x;
    if (i < n) o[i] = (half_t)a[i];
}

// ---------------------------------------------------------------------------
// LayerNorm (per-token), fp32 in -> fp16 out.  One 256-thread block per token.
// ---------------------------------------------------------------------------
__global__ __launch_bounds__(256)
void layernorm_f16_kernel(const float* __restrict__ in, const float* __restrict__ g,
                          const float* __restrict__ bta, half_t* __restrict__ out) {
    const int tok = blockIdx.x;
    const int tid = threadIdx.x;
    const float* row = in + (size_t)tok * D_MODEL;
    __shared__ float red[256];

    float v[3];
    float s = 0.f, sq = 0.f;
    #pragma unroll
    for (int i = 0; i < 3; ++i) {
        v[i] = row[tid + i * 256];
        s  += v[i];
        sq += v[i] * v[i];
    }
    red[tid] = s; __syncthreads();
    for (int off = 128; off > 0; off >>= 1) { if (tid < off) red[tid] += red[tid + off]; __syncthreads(); }
    const float mean = red[0] / (float)D_MODEL;
    __syncthreads();
    red[tid] = sq; __syncthreads();
    for (int off = 128; off > 0; off >>= 1) { if (tid < off) red[tid] += red[tid + off]; __syncthreads(); }
    const float var = red[0] / (float)D_MODEL - mean * mean;
    const float rstd = rsqrtf(var + 1e-5f);

    #pragma unroll
    for (int i = 0; i < 3; ++i) {
        int c = tid + i * 256;
        out[(size_t)tok * D_MODEL + c] = (half_t)((v[i] - mean) * rstd * g[c] + bta[c]);
    }
}

// copy CLS rows of x into xin
__global__ void copy_cls_kernel(const float* __restrict__ x, float* __restrict__ xin) {
    int b = blockIdx.x;
    for (int c = threadIdx.x; c < D_MODEL; c += blockDim.x)
        xin[(size_t)b * N_TOK * D_MODEL + c] = x[(size_t)b * N_TOK * D_MODEL + c];
}

// ---------------------------------------------------------------------------
// Generic tiled WMMA GEMM:  C[M,N] = epilogue(A[M,K] @ W[K,N])
// A, W fp16 row-major.  Block tile 128x64, 8 wave32 waves (4x2), each wave
// computes a 32x32 tile = 2x2 v_wmma_f32_16x16x32_f16.
// Interior A tiles are staged with global_load_async_to_lds_b128; boundary
// tiles and the transposed W tile use the guarded VGPR path.
// ---------------------------------------------------------------------------
template<bool GELU, bool F16OUT, bool ACCUM>
__global__ __launch_bounds__(256)
void gemm_f16_wmma(const half_t* __restrict__ A, int lda,
                   const half_t* __restrict__ W, int ldw,
                   const float* __restrict__ bias,
                   const float* __restrict__ gamma,
                   const float* __restrict__ residual, int ldr,
                   float* __restrict__ Cf, half_t* __restrict__ Ch, int ldc,
                   int M, int N, int K) {
    __shared__ __align__(16) half_t As[128][32];
    __shared__ __align__(16) half_t Bt[64][32];   // transposed: [n][k]

    const int tid  = threadIdx.x;
    const int lane = tid & 31;
    const int wave = tid >> 5;
    const int wm = (wave >> 1) * 32;
    const int wn = (wave & 1) * 32;
    const int m0 = blockIdx.y * 128;
    const int n0 = blockIdx.x * 64;

    v8f acc[2][2];
    #pragma unroll
    for (int i = 0; i < 2; ++i)
        #pragma unroll
        for (int j = 0; j < 2; ++j) { v8f z = {}; acc[i][j] = z; }

    const int arow = tid >> 1;          // 0..127
    const int acs  = (tid & 1) * 16;    // 0 or 16
    const int wk_  = tid >> 3;          // 0..31
    const int wns  = (tid & 7) * 8;     // 0..56

    for (int kk = 0; kk < K; kk += 32) {
        // --- stage A tile (128 x 32) ---
        if ((m0 + 128 <= M) && (kk + 32 <= K)) {
            const half_t* gp = A + (size_t)(m0 + arow) * lda + kk + acs;
            async_copy_b128(gp,     &As[arow][acs]);
            async_copy_b128(gp + 8, &As[arow][acs + 8]);
        } else {
            int grow = m0 + arow;
            uint4 d0 = {0,0,0,0}, d1 = {0,0,0,0};
            if (grow < M) {
                if (kk + acs     < K) d0 = *reinterpret_cast<const uint4*>(A + (size_t)grow * lda + kk + acs);
                if (kk + acs + 8 < K) d1 = *reinterpret_cast<const uint4*>(A + (size_t)grow * lda + kk + acs + 8);
            }
            *reinterpret_cast<uint4*>(&As[arow][acs])     = d0;
            *reinterpret_cast<uint4*>(&As[arow][acs + 8]) = d1;
        }
        // --- stage W tile (32 x 64) transposed into Bt ---
        {
            union { uint4 u; half_t h[8]; } t; t.u = {0,0,0,0};
            if ((kk + wk_ < K) && (n0 + wns < N))
                t.u = *reinterpret_cast<const uint4*>(W + (size_t)(kk + wk_) * ldw + n0 + wns);
            #pragma unroll
            for (int j = 0; j < 8; ++j) Bt[wns + j][wk_] = t.h[j];
        }
        async_wait0();
        __syncthreads();

        // --- build fragments & WMMA ---
        v16h afr[2], bfr[2];
        #pragma unroll
        for (int mt = 0; mt < 2; ++mt) {
            int r  = wm + mt * 16 + (lane & 15);
            int kb = (lane < 16) ? 0 : 8;
            v8h lo = *reinterpret_cast<const v8h*>(&As[r][kb]);
            v8h hi = *reinterpret_cast<const v8h*>(&As[r][kb + 16]);
            #pragma unroll
            for (int i = 0; i < 8; ++i) { afr[mt][i] = lo[i]; afr[mt][8 + i] = hi[i]; }
        }
        #pragma unroll
        for (int nt = 0; nt < 2; ++nt) {
            int c  = wn + nt * 16 + (lane & 15);
            int kb = (lane < 16) ? 0 : 8;
            v8h lo = *reinterpret_cast<const v8h*>(&Bt[c][kb]);
            v8h hi = *reinterpret_cast<const v8h*>(&Bt[c][kb + 16]);
            #pragma unroll
            for (int i = 0; i < 8; ++i) { bfr[nt][i] = lo[i]; bfr[nt][8 + i] = hi[i]; }
        }
        #pragma unroll
        for (int mt = 0; mt < 2; ++mt)
            #pragma unroll
            for (int nt = 0; nt < 2; ++nt)
                acc[mt][nt] = __builtin_amdgcn_wmma_f32_16x16x32_f16(
                    false, afr[mt], false, bfr[nt], (short)0, acc[mt][nt], false, false);
        __syncthreads();
    }

    // --- epilogue ---
    #pragma unroll
    for (int mt = 0; mt < 2; ++mt)
        #pragma unroll
        for (int nt = 0; nt < 2; ++nt)
            #pragma unroll
            for (int r = 0; r < 8; ++r) {
                int grow = m0 + wm + mt * 16 + r + ((lane >> 4) * 8);
                int gcol = n0 + wn + nt * 16 + (lane & 15);
                if (grow < M && gcol < N) {
                    float v = acc[mt][nt][r];
                    if (bias)     v += bias[gcol];
                    if (GELU)     v  = gelu_f(v);
                    if (gamma)    v *= gamma[gcol];
                    if (residual) v += residual[(size_t)grow * ldr + gcol];
                    if constexpr (F16OUT)     Ch[(size_t)grow * ldc + gcol]  = (half_t)v;
                    else if constexpr (ACCUM) Cf[(size_t)grow * ldc + gcol] += v;
                    else                      Cf[(size_t)grow * ldc + gcol]  = v;
                }
            }
}

// ---------------------------------------------------------------------------
// Attention scores: S[b,h,q,k] = scale * q[b,q,h,:] . k[b,k,h,:]
// q,k are fp16 [B*N_TOK, 768] with head offset h*64.  Written straight into
// the fp32 attn region of d_out (pre-softmax).  Interior q/k tiles staged
// with async global->LDS copies (k tile needs no transpose).
// ---------------------------------------------------------------------------
__global__ __launch_bounds__(256)
void attn_scores_kernel(const half_t* __restrict__ q, const half_t* __restrict__ k,
                        float* __restrict__ attn, float scale) {
    __shared__ __align__(16) half_t As[128][32];
    __shared__ __align__(16) half_t Bt[64][32];

    const int tid  = threadIdx.x;
    const int lane = tid & 31;
    const int wave = tid >> 5;
    const int wm = (wave >> 1) * 32, wn = (wave & 1) * 32;
    const int m0 = blockIdx.y * 128, n0 = blockIdx.x * 64;
    const int bh = blockIdx.z, b = bh / N_HEADS, h = bh % N_HEADS;

    const half_t* qb = q + (size_t)b * N_TOK * D_MODEL + h * HEAD_D;
    const half_t* kb = k + (size_t)b * N_TOK * D_MODEL + h * HEAD_D;
    float* ab = attn + (size_t)bh * N_TOK * N_TOK;

    v8f acc[2][2];
    #pragma unroll
    for (int i = 0; i < 2; ++i)
        #pragma unroll
        for (int j = 0; j < 2; ++j) { v8f z = {}; acc[i][j] = z; }

    const int arow = tid >> 1, acs = (tid & 1) * 16;
    const int btok = tid >> 2, bseg = (tid & 3) * 8;

    for (int kk = 0; kk < HEAD_D; kk += 32) {
        // q tile
        if (m0 + 128 <= N_TOK) {
            const half_t* gp = qb + (size_t)(m0 + arow) * D_MODEL + kk + acs;
            async_copy_b128(gp,     &As[arow][acs]);
            async_copy_b128(gp + 8, &As[arow][acs + 8]);
        } else {
            int grow = m0 + arow;
            uint4 d0 = {0,0,0,0}, d1 = {0,0,0,0};
            if (grow < N_TOK) {
                d0 = *reinterpret_cast<const uint4*>(qb + (size_t)grow * D_MODEL + kk + acs);
                d1 = *reinterpret_cast<const uint4*>(qb + (size_t)grow * D_MODEL + kk + acs + 8);
            }
            *reinterpret_cast<uint4*>(&As[arow][acs])     = d0;
            *reinterpret_cast<uint4*>(&As[arow][acs + 8]) = d1;
        }
        // k tile (already [n][kdim] layout)
        if (n0 + 64 <= N_TOK) {
            async_copy_b128(kb + (size_t)(n0 + btok) * D_MODEL + kk + bseg, &Bt[btok][bseg]);
        } else {
            uint4 d = {0,0,0,0};
            if (n0 + btok < N_TOK)
                d = *reinterpret_cast<const uint4*>(kb + (size_t)(n0 + btok) * D_MODEL + kk + bseg);
            *reinterpret_cast<uint4*>(&Bt[btok][bseg]) = d;
        }
        async_wait0();
        __syncthreads();

        v16h afr[2], bfr[2];
        #pragma unroll
        for (int mt = 0; mt < 2; ++mt) {
            int r = wm + mt * 16 + (lane & 15);
            int kbo = (lane < 16) ? 0 : 8;
            v8h lo = *reinterpret_cast<const v8h*>(&As[r][kbo]);
            v8h hi = *reinterpret_cast<const v8h*>(&As[r][kbo + 16]);
            #pragma unroll
            for (int i = 0; i < 8; ++i) { afr[mt][i] = lo[i]; afr[mt][8 + i] = hi[i]; }
        }
        #pragma unroll
        for (int nt = 0; nt < 2; ++nt) {
            int c = wn + nt * 16 + (lane & 15);
            int kbo = (lane < 16) ? 0 : 8;
            v8h lo = *reinterpret_cast<const v8h*>(&Bt[c][kbo]);
            v8h hi = *reinterpret_cast<const v8h*>(&Bt[c][kbo + 16]);
            #pragma unroll
            for (int i = 0; i < 8; ++i) { bfr[nt][i] = lo[i]; bfr[nt][8 + i] = hi[i]; }
        }
        #pragma unroll
        for (int mt = 0; mt < 2; ++mt)
            #pragma unroll
            for (int nt = 0; nt < 2; ++nt)
                acc[mt][nt] = __builtin_amdgcn_wmma_f32_16x16x32_f16(
                    false, afr[mt], false, bfr[nt], (short)0, acc[mt][nt], false, false);
        __syncthreads();
    }

    #pragma unroll
    for (int mt = 0; mt < 2; ++mt)
        #pragma unroll
        for (int nt = 0; nt < 2; ++nt)
            #pragma unroll
            for (int r = 0; r < 8; ++r) {
                int grow = m0 + wm + mt * 16 + r + ((lane >> 4) * 8);
                int gcol = n0 + wn + nt * 16 + (lane & 15);
                if (grow < N_TOK && gcol < N_TOK)
                    ab[(size_t)grow * N_TOK + gcol] = scale * acc[mt][nt][r];
            }
}

// ---------------------------------------------------------------------------
// In-place row softmax over N_TOK elements.  One 256-thread block per row.
// ---------------------------------------------------------------------------
__global__ __launch_bounds__(256)
void softmax_row_kernel(float* __restrict__ attn) {
    float* p = attn + (size_t)blockIdx.x * N_TOK;
    const int tid = threadIdx.x;
    __shared__ float red[256];

    float mx = -1e30f;
    for (int i = tid; i < N_TOK; i += 256) mx = fmaxf(mx, p[i]);
    red[tid] = mx; __syncthreads();
    for (int off = 128; off > 0; off >>= 1) { if (tid < off) red[tid] = fmaxf(red[tid], red[tid + off]); __syncthreads(); }
    mx = red[0]; __syncthreads();

    float s = 0.f;
    for (int i = tid; i < N_TOK; i += 256) s += __expf(p[i] - mx);
    red[tid] = s; __syncthreads();
    for (int off = 128; off > 0; off >>= 1) { if (tid < off) red[tid] += red[tid + off]; __syncthreads(); }
    const float inv = 1.0f / red[0];

    for (int i = tid; i < N_TOK; i += 256) p[i] = __expf(p[i] - mx) * inv;
}

// ---------------------------------------------------------------------------
// attn @ V :  out[b, n, h*64+d] = sum_k attn[b,h,n,k] * v[b,k,h*64+d]
// attn fp32 (converted to f16 while staging into LDS), v fp16, out fp16.
// ---------------------------------------------------------------------------
__global__ __launch_bounds__(256)
void attn_av_kernel(const float* __restrict__ attn, const half_t* __restrict__ v,
                    half_t* __restrict__ out) {
    __shared__ __align__(16) half_t As[128][32];
    __shared__ __align__(16) half_t Bt[64][32];

    const int tid  = threadIdx.x;
    const int lane = tid & 31;
    const int wave = tid >> 5;
    const int wm = (wave >> 1) * 32, wn = (wave & 1) * 32;
    const int m0 = blockIdx.y * 128;
    const int bh = blockIdx.z, b = bh / N_HEADS, h = bh % N_HEADS;

    const float*  ab = attn + (size_t)bh * N_TOK * N_TOK;
    const half_t* vb = v   + (size_t)b * N_TOK * D_MODEL + h * HEAD_D;
    half_t*       ob = out + (size_t)b * N_TOK * D_MODEL + h * HEAD_D;

    v8f acc[2][2];
    #pragma unroll
    for (int i = 0; i < 2; ++i)
        #pragma unroll
        for (int j = 0; j < 2; ++j) { v8f z = {}; acc[i][j] = z; }

    const int arow = tid >> 1, acs = (tid & 1) * 16;
    const int btok = tid >> 3, bseg = (tid & 7) * 8;

    for (int kk = 0; kk < N_TOK; kk += 32) {
        {
            int grow = m0 + arow;
            #pragma unroll
            for (int j = 0; j < 16; ++j) {
                int col = kk + acs + j;
                float val = (grow < N_TOK && col < N_TOK) ? ab[(size_t)grow * N_TOK + col] : 0.f;
                As[arow][acs + j] = (half_t)val;
            }
        }
        {
            union { uint4 u; half_t hh[8]; } t; t.u = {0,0,0,0};
            if (kk + btok < N_TOK)
                t.u = *reinterpret_cast<const uint4*>(vb + (size_t)(kk + btok) * D_MODEL + bseg);
            #pragma unroll
            for (int j = 0; j < 8; ++j) Bt[bseg + j][btok] = t.hh[j];
        }
        __syncthreads();

        v16h afr[2], bfr[2];
        #pragma unroll
        for (int mt = 0; mt < 2; ++mt) {
            int r = wm + mt * 16 + (lane & 15);
            int kbo = (lane < 16) ? 0 : 8;
            v8h lo = *reinterpret_cast<const v8h*>(&As[r][kbo]);
            v8h hi = *reinterpret_cast<const v8h*>(&As[r][kbo + 16]);
            #pragma unroll
            for (int i = 0; i < 8; ++i) { afr[mt][i] = lo[i]; afr[mt][8 + i] = hi[i]; }
        }
        #pragma unroll
        for (int nt = 0; nt < 2; ++nt) {
            int c = wn + nt * 16 + (lane & 15);
            int kbo = (lane < 16) ? 0 : 8;
            v8h lo = *reinterpret_cast<const v8h*>(&Bt[c][kbo]);
            v8h hi = *reinterpret_cast<const v8h*>(&Bt[c][kbo + 16]);
            #pragma unroll
            for (int i = 0; i < 8; ++i) { bfr[nt][i] = lo[i]; bfr[nt][8 + i] = hi[i]; }
        }
        #pragma unroll
        for (int mt = 0; mt < 2; ++mt)
            #pragma unroll
            for (int nt = 0; nt < 2; ++nt)
                acc[mt][nt] = __builtin_amdgcn_wmma_f32_16x16x32_f16(
                    false, afr[mt], false, bfr[nt], (short)0, acc[mt][nt], false, false);
        __syncthreads();
    }

    #pragma unroll
    for (int mt = 0; mt < 2; ++mt)
        #pragma unroll
        for (int nt = 0; nt < 2; ++nt)
            #pragma unroll
            for (int r = 0; r < 8; ++r) {
                int grow = m0 + wm + mt * 16 + r + ((lane >> 4) * 8);
                int gcol = wn + nt * 16 + (lane & 15);   // < 64 always
                if (grow < N_TOK)
                    ob[(size_t)grow * D_MODEL + gcol] = (half_t)acc[mt][nt][r];
            }
}

// ---------------------------------------------------------------------------
// Token selector: head-averaged CLS attention over patch tokens, top-256
// (kept in original order), cutoff mask, gather into [B, 257, 768] output.
// One 1024-thread block per batch element.
// ---------------------------------------------------------------------------
__global__ __launch_bounds__(1024)
void selector_kernel(const float* __restrict__ attn, const float* __restrict__ xfull,
                     float* __restrict__ xsel) {
    const int b = blockIdx.x, tid = threadIdx.x;
    __shared__ float sval[1024];
    __shared__ int   sidx[1024];
    __shared__ float tsc[256];
    __shared__ int   tix[256];
    __shared__ float cutoff_s;

    float s = 0.f;
    for (int h = 0; h < N_HEADS; ++h)
        s += attn[((size_t)(b * N_HEADS + h) * N_TOK + 0) * N_TOK + 1 + tid];
    sval[tid] = s * (1.0f / N_HEADS);
    sidx[tid] = tid;
    __syncthreads();

    // bitonic sort, descending by score
    for (int k = 2; k <= 1024; k <<= 1)
        for (int j = k >> 1; j > 0; j >>= 1) {
            int ixj = tid ^ j;
            if (ixj > tid) {
                bool desc = ((tid & k) == 0);
                float a = sval[tid], c = sval[ixj];
                bool sw = desc ? (a < c) : (a > c);
                if (sw) {
                    sval[tid] = c; sval[ixj] = a;
                    int t2 = sidx[tid]; sidx[tid] = sidx[ixj]; sidx[ixj] = t2;
                }
            }
            __syncthreads();
        }

    if (tid == 0) cutoff_s = 0.15f * sval[0];
    if (tid < 256) { tix[tid] = sidx[tid]; tsc[tid] = sval[tid]; }
    __syncthreads();

    // bitonic sort top-256 ascending by original index
    for (int k = 2; k <= 256; k <<= 1)
        for (int j = k >> 1; j > 0; j >>= 1) {
            if (tid < 256) {
                int ixj = tid ^ j;
                if (ixj > tid) {
                    bool asc = ((tid & k) == 0);
                    int a = tix[tid], c = tix[ixj];
                    bool sw = asc ? (a > c) : (a < c);
                    if (sw) {
                        tix[tid] = c; tix[ixj] = a;
                        float t2 = tsc[tid]; tsc[tid] = tsc[ixj]; tsc[ixj] = t2;
                    }
                }
            }
            __syncthreads();
        }
    const float cut = cutoff_s;

    for (int c = tid; c < D_MODEL; c += 1024)
        xsel[((size_t)b * N_SEL) * D_MODEL + c] = xfull[((size_t)b * N_TOK) * D_MODEL + c];

    for (int i = tid; i < 256 * D_MODEL; i += 1024) {
        int srow = i / D_MODEL, c = i % D_MODEL;
        int tok = tix[srow];
        float keep = (tsc[srow] >= cut) ? 1.0f : 0.0f;
        xsel[((size_t)(b * N_SEL + 1 + srow)) * D_MODEL + c] =
            xfull[((size_t)(b * N_TOK + 1 + tok)) * D_MODEL + c] * keep;
    }
}

// ---------------------------------------------------------------------------
// host launcher
// ---------------------------------------------------------------------------
extern "C" void kernel_launch(void* const* d_in, const int* in_sizes, int n_in,
                              void* d_out, int out_size, void* d_ws, size_t ws_size,
                              hipStream_t stream) {
    (void)in_sizes; (void)n_in; (void)out_size; (void)ws_size;

    const float* x_FMAG  = (const float*)d_in[1];
    const float* x_in    = (const float*)d_in[2];
    const float* z_in    = (const float*)d_in[3];
    const float* ln1_g   = (const float*)d_in[4];
    const float* ln1_b   = (const float*)d_in[5];
    const float* ln2_g   = (const float*)d_in[6];
    const float* ln2_b   = (const float*)d_in[7];
    const float* wq      = (const float*)d_in[8];
    const float* wk      = (const float*)d_in[9];
    const float* wv      = (const float*)d_in[10];
    const float* wo      = (const float*)d_in[11];
    const float* bo      = (const float*)d_in[12];
    const float* mlp_w1  = (const float*)d_in[13];
    const float* mlp_b1  = (const float*)d_in[14];
    const float* mlp_w2  = (const float*)d_in[15];
    const float* mlp_b2  = (const float*)d_in[16];
    const float* st_wd   = (const float*)d_in[17];
    const float* st_bd   = (const float*)d_in[18];
    const float* up_w    = (const float*)d_in[21];
    const float* up_b    = (const float*)d_in[22];
    const float* bi_wd   = (const float*)d_in[25];
    const float* bi_bd   = (const float*)d_in[26];
    const float* bi_wm   = (const float*)d_in[27];
    const float* bi_bm   = (const float*)d_in[28];
    const float* bi_wu   = (const float*)d_in[29];
    const float* bi_bu   = (const float*)d_in[30];
    const float* gamma_1 = (const float*)d_in[31];
    const float* gamma_2 = (const float*)d_in[32];

    // output layout (flat f32): x[4,257,768] | attn[4,12,1025,1025] | z | i_attn
    const size_t SZ_X    = (size_t)BATCH * N_SEL * D_MODEL;          // 789504
    const size_t SZ_ATTN = (size_t)BATCH * N_HEADS * N_TOK * N_TOK;  // 50430000
    float* out_x     = (float*)d_out;
    float* out_attn  = out_x + SZ_X;
    float* out_z     = out_attn + SZ_ATTN;
    float* out_iattn = out_z + SZ_X;

    // workspace carve-up
    char* ws = (char*)d_ws;
    size_t cur = 0;
    auto alloc = [&](size_t bytes) -> char* {
        char* p = ws + cur;
        cur = (cur + bytes + 255) & ~(size_t)255;
        return p;
    };
    const int M_ALL = BATCH * N_TOK;   // 4100
    const int M_FM  = BATCH * 1024;    // 4096
    const int M_SEL = BATCH * N_SEL;   // 1028

    half_t* wq16   = (half_t*)alloc((size_t)768 * 768 * 2);
    half_t* wk16   = (half_t*)alloc((size_t)768 * 768 * 2);
    half_t* wv16   = (half_t*)alloc((size_t)768 * 768 * 2);
    half_t* wo16   = (half_t*)alloc((size_t)768 * 768 * 2);
    half_t* w1_16  = (half_t*)alloc((size_t)768 * 3072 * 2);
    half_t* w2_16  = (half_t*)alloc((size_t)3072 * 768 * 2);
    half_t* stwd16 = (half_t*)alloc((size_t)768 * 96 * 2);
    half_t* upw16  = (half_t*)alloc((size_t)96 * 768 * 2);
    half_t* biwd16 = (half_t*)alloc((size_t)768 * 16 * 2);
    half_t* biwm16 = (half_t*)alloc((size_t)16 * 16 * 2);
    half_t* biwu16 = (half_t*)alloc((size_t)16 * 768 * 2);

    half_t* lnfm16  = (half_t*)alloc((size_t)M_FM * 768 * 2);
    half_t* fmagh16 = (half_t*)alloc((size_t)M_FM * 96 * 2);
    float*  xin     = (float*) alloc((size_t)M_ALL * 768 * 4);
    half_t* lnx16   = (half_t*)alloc((size_t)M_ALL * 768 * 2);
    half_t* q16     = (half_t*)alloc((size_t)M_ALL * 768 * 2);
    half_t* k16     = (half_t*)alloc((size_t)M_ALL * 768 * 2);
    half_t* v16     = (half_t*)alloc((size_t)M_ALL * 768 * 2);
    half_t* ao16    = (half_t*)alloc((size_t)M_ALL * 768 * 2);
    float*  x1      = (float*) alloc((size_t)M_ALL * 768 * 4);
    float*  z1      = (float*) alloc((size_t)M_ALL * 768 * 4);
    half_t* lnxs16  = (half_t*)alloc((size_t)M_SEL * 768 * 2);
    half_t* lnzs16  = (half_t*)alloc((size_t)M_SEL * 768 * 2);
    half_t* h1_16   = (half_t*)alloc((size_t)M_SEL * 3072 * 2);
    half_t* a1_16   = (half_t*)alloc((size_t)M_SEL * 16 * 2);
    half_t* a2_16   = (half_t*)alloc((size_t)M_SEL * 16 * 2);

    const dim3 blk(256);
    auto conv = [&](const float* a, half_t* o, int n) {
        f32_to_f16_kernel<<<dim3((n + 255) / 256), blk, 0, stream>>>(a, o, n);
    };
    auto ggrid = [](int Mv, int Nv) { return dim3((Nv + 63) / 64, (Mv + 127) / 128, 1); };

    // ---- weight conversion ----
    conv(wq, wq16, 768 * 768);   conv(wk, wk16, 768 * 768);
    conv(wv, wv16, 768 * 768);   conv(wo, wo16, 768 * 768);
    conv(mlp_w1, w1_16, 768 * 3072);  conv(mlp_w2, w2_16, 3072 * 768);
    conv(st_wd, stwd16, 768 * 96);    conv(up_w, upw16, 96 * 768);
    conv(bi_wd, biwd16, 768 * 16);    conv(bi_wm, biwm16, 16 * 16);
    conv(bi_wu, biwu16, 16 * 768);

    // ---- FMAG fusion: xin = [x_cls, gelu(ln2(x_FMAG)@st_wd+b)@up_w+b + x_patch] ----
    layernorm_f16_kernel<<<dim3(M_FM), blk, 0, stream>>>(x_FMAG, ln2_g, ln2_b, lnfm16);
    gemm_f16_wmma<true, true, false><<<ggrid(M_FM, 96), blk, 0, stream>>>(
        lnfm16, 768, stwd16, 96, st_bd, nullptr, nullptr, 0,
        nullptr, fmagh16, 96, M_FM, 96, 768);
    for (int b = 0; b < BATCH; ++b)
        gemm_f16_wmma<false, false, false><<<ggrid(1024, 768), blk, 0, stream>>>(
            fmagh16 + (size_t)b * 1024 * 96, 96, upw16, 768, up_b, nullptr,
            x_in + ((size_t)b * N_TOK + 1) * 768, 768,
            xin + ((size_t)b * N_TOK + 1) * 768, nullptr, 768, 1024, 768, 96);
    copy_cls_kernel<<<dim3(BATCH), blk, 0, stream>>>(x_in, xin);

    // ---- x path: MHA ----
    layernorm_f16_kernel<<<dim3(M_ALL), blk, 0, stream>>>(xin, ln1_g, ln1_b, lnx16);
    gemm_f16_wmma<false, true, false><<<ggrid(M_ALL, 768), blk, 0, stream>>>(
        lnx16, 768, wq16, 768, nullptr, nullptr, nullptr, 0, nullptr, q16, 768, M_ALL, 768, 768);
    gemm_f16_wmma<false, true, false><<<ggrid(M_ALL, 768), blk, 0, stream>>>(
        lnx16, 768, wk16, 768, nullptr, nullptr, nullptr, 0, nullptr, k16, 768, M_ALL, 768, 768);
    gemm_f16_wmma<false, true, false><<<ggrid(M_ALL, 768), blk, 0, stream>>>(
        lnx16, 768, wv16, 768, nullptr, nullptr, nullptr, 0, nullptr, v16, 768, M_ALL, 768, 768);

    const dim3 sgrid((N_TOK + 63) / 64, (N_TOK + 127) / 128, BATCH * N_HEADS);
    const dim3 avgrid(1, (N_TOK + 127) / 128, BATCH * N_HEADS);
    attn_scores_kernel<<<sgrid, blk, 0, stream>>>(q16, k16, out_attn, 0.125f);
    softmax_row_kernel<<<dim3(BATCH * N_HEADS * N_TOK), blk, 0, stream>>>(out_attn);
    attn_av_kernel<<<avgrid, blk, 0, stream>>>(out_attn, v16, ao16);
    gemm_f16_wmma<false, false, false><<<ggrid(M_ALL, 768), blk, 0, stream>>>(
        ao16, 768, wo16, 768, bo, gamma_1, xin, 768, x1, nullptr, 768, M_ALL, 768, 768);
    selector_kernel<<<dim3(BATCH), dim3(1024), 0, stream>>>(out_attn, x1, out_x);

    // ---- z path: MHA (reuse buffers) ----
    layernorm_f16_kernel<<<dim3(M_ALL), blk, 0, stream>>>(z_in, ln1_g, ln1_b, lnx16);
    gemm_f16_wmma<false, true, false><<<ggrid(M_ALL, 768), blk, 0, stream>>>(
        lnx16, 768, wq16, 768, nullptr, nullptr, nullptr, 0, nullptr, q16, 768, M_ALL, 768, 768);
    gemm_f16_wmma<false, true, false><<<ggrid(M_ALL, 768), blk, 0, stream>>>(
        lnx16, 768, wk16, 768, nullptr, nullptr, nullptr, 0, nullptr, k16, 768, M_ALL, 768, 768);
    gemm_f16_wmma<false, true, false><<<ggrid(M_ALL, 768), blk, 0, stream>>>(
        lnx16, 768, wv16, 768, nullptr, nullptr, nullptr, 0, nullptr, v16, 768, M_ALL, 768, 768);
    attn_scores_kernel<<<sgrid, blk, 0, stream>>>(q16, k16, out_iattn, 0.125f);
    softmax_row_kernel<<<dim3(BATCH * N_HEADS * N_TOK), blk, 0, stream>>>(out_iattn);
    attn_av_kernel<<<avgrid, blk, 0, stream>>>(out_iattn, v16, ao16);
    gemm_f16_wmma<false, false, false><<<ggrid(M_ALL, 768), blk, 0, stream>>>(
        ao16, 768, wo16, 768, bo, gamma_1, z_in, 768, z1, nullptr, 768, M_ALL, 768, 768);
    selector_kernel<<<dim3(BATCH), dim3(1024), 0, stream>>>(out_iattn, z1, out_z);

    // ---- MLP + bi-adapter cross terms ----
    layernorm_f16_kernel<<<dim3(M_SEL), blk, 0, stream>>>(out_x, ln2_g, ln2_b, lnxs16);
    layernorm_f16_kernel<<<dim3(M_SEL), blk, 0, stream>>>(out_z, ln2_g, ln2_b, lnzs16);

    // x += gamma2*mlp(ln2(x)) + bi(ln2(z))
    gemm_f16_wmma<true, true, false><<<ggrid(M_SEL, 3072), blk, 0, stream>>>(
        lnxs16, 768, w1_16, 3072, mlp_b1, nullptr, nullptr, 0, nullptr, h1_16, 3072, M_SEL, 3072, 768);
    gemm_f16_wmma<false, false, true><<<ggrid(M_SEL, 768), blk, 0, stream>>>(
        h1_16, 3072, w2_16, 768, mlp_b2, gamma_2, nullptr, 0, out_x, nullptr, 768, M_SEL, 768, 3072);
    gemm_f16_wmma<false, true, false><<<ggrid(M_SEL, 16), blk, 0, stream>>>(
        lnzs16, 768, biwd16, 16, bi_bd, nullptr, nullptr, 0, nullptr, a1_16, 16, M_SEL, 16, 768);
    gemm_f16_wmma<false, true, false><<<ggrid(M_SEL, 16), blk, 0, stream>>>(
        a1_16, 16, biwm16, 16, bi_bm, nullptr, nullptr, 0, nullptr, a2_16, 16, M_SEL, 16, 16);
    gemm_f16_wmma<false, false, true><<<ggrid(M_SEL, 768), blk, 0, stream>>>(
        a2_16, 16, biwu16, 768, bi_bu, nullptr, nullptr, 0, out_x, nullptr, 768, M_SEL, 768, 16);

    // z += gamma2*mlp(ln2(z)) + bi(ln2(x_ori))
    gemm_f16_wmma<true, true, false><<<ggrid(M_SEL, 3072), blk, 0, stream>>>(
        lnzs16, 768, w1_16, 3072, mlp_b1, nullptr, nullptr, 0, nullptr, h1_16, 3072, M_SEL, 3072, 768);
    gemm_f16_wmma<false, false, true><<<ggrid(M_SEL, 768), blk, 0, stream>>>(
        h1_16, 3072, w2_16, 768, mlp_b2, gamma_2, nullptr, 0, out_z, nullptr, 768, M_SEL, 768, 3072);
    gemm_f16_wmma<false, true, false><<<ggrid(M_SEL, 16), blk, 0, stream>>>(
        lnxs16, 768, biwd16, 16, bi_bd, nullptr, nullptr, 0, nullptr, a1_16, 16, M_SEL, 16, 768);
    gemm_f16_wmma<false, true, false><<<ggrid(M_SEL, 16), blk, 0, stream>>>(
        a1_16, 16, biwm16, 16, bi_bm, nullptr, nullptr, 0, nullptr, a2_16, 16, M_SEL, 16, 16);
    gemm_f16_wmma<false, false, true><<<ggrid(M_SEL, 768), blk, 0, stream>>>(
        a2_16, 16, biwu16, 768, bi_bu, nullptr, nullptr, 0, out_z, nullptr, 768, M_SEL, 768, 16);
}
